// CAM_Module_120259084308
// MI455X (gfx1250) — matrix operand order repaired and checked by reading, not compiled, not run on hardware
//
#include <hip/hip_runtime.h>
#include <hip/hip_bf16.h>

typedef __attribute__((ext_vector_type(16))) __bf16          bf16x16;
typedef __attribute__((ext_vector_type(8)))  float           f32x8;
typedef __attribute__((ext_vector_type(8)))  unsigned short  u16x8;

#define BATCH 16
#define CCH   512     // channels
#define NSP   4096    // spatial H*W
#define TILE  128
#define KSTEP 32
#define A_ST  40      // bf16 elems per LDS row (32 used + 8 pad) = 80B, conflict-free
#define BT_ST 20      // dwords per transposed LDS row (16 used + 4 pad) = 80B

union FragCvt { uint4 u[2]; bf16x16 v; };

#if __has_builtin(__builtin_amdgcn_tensor_load_to_lds) && __has_builtin(__builtin_amdgcn_s_wait_tensorcnt)
#define USE_TDM 1
#else
#define USE_TDM 0
#endif

#if USE_TDM
typedef __attribute__((ext_vector_type(4))) unsigned int u32x4;
typedef __attribute__((ext_vector_type(8))) int          i32x8;
typedef __attribute__((ext_vector_type(4))) int          i32x4;

// 2D tile DMA: global (row-major, stride0_elems bf16 per row) -> LDS, with
// 16B padding appended after every 64B row so LDS row pitch becomes 80B (A_ST/BT_ST).
__device__ __forceinline__ void tdm_load_tile_2d(const __bf16* gsrc, unsigned lds_byte_addr,
                                                 unsigned tile_d0, unsigned tile_d1,
                                                 unsigned stride0_elems) {
  unsigned long long ga = (unsigned long long)(size_t)gsrc;
  u32x4 g0;
  g0.x = 1u;                                    // count=1: one valid user descriptor
  g0.y = lds_byte_addr;                         // D#.lds_addr
  g0.z = (unsigned)ga;                          // D#.global_addr[31:0]
  g0.w = (unsigned)((ga >> 32) & 0x01ffffffu)   // D#.global_addr[56:32]
       | (2u << 30);                            // D#.type = 2 ("image")
  i32x8 g1;
  g1[0] = (int)((1u << 16)                      // data_size = 2 bytes
              | (1u << 20)                      // pad_enable
              | (3u << 22)                      // pad_interval: every 16 DWORDs (64B)
              | (3u << 25));                    // pad_amount: 4 DWORDs (16B)
  g1[1] = (int)((tile_d0 & 0xffffu) << 16);                                   // tensor_dim0[15:0]
  g1[2] = (int)(((tile_d0 >> 16) & 0xffffu) | ((tile_d1 & 0xffffu) << 16));   // dim0 hi | dim1 lo
  g1[3] = (int)(((tile_d1 >> 16) & 0xffffu) | ((tile_d0 & 0xffffu) << 16));   // dim1 hi | tile_dim0
  g1[4] = (int)(tile_d1 & 0xffffu);                                           // tile_dim1 (tile_dim2=0)
  g1[5] = (int)stride0_elems;                                                 // tensor_dim0_stride lo32
  g1[6] = 0;                                                                  // stride hi / dim1_stride
  g1[7] = 0;
  i32x4 z4 = {0, 0, 0, 0};                      // groups 2/3: unused (2D tensor)
  i32x8 z8 = {0, 0, 0, 0, 0, 0, 0, 0};          // extra group (clang-23 6-arg form)
  __builtin_amdgcn_tensor_load_to_lds(g0, g1, z4, z4, z8, 0);
}
#endif

// ---------------------------------------------------------------- convert
__global__ __launch_bounds__(256)
void cam_convert(const float* __restrict__ x, __bf16* __restrict__ v) {
  size_t i = (size_t)blockIdx.x * blockDim.x + threadIdx.x;
  const float4* xin = (const float4*)x + i * 2;
  float4 a = xin[0];
  float4 b = xin[1];
  union { __bf16 h[8]; uint4 u; } o;
  o.h[0] = (__bf16)a.x; o.h[1] = (__bf16)a.y; o.h[2] = (__bf16)a.z; o.h[3] = (__bf16)a.w;
  o.h[4] = (__bf16)b.x; o.h[5] = (__bf16)b.y; o.h[6] = (__bf16)b.z; o.h[7] = (__bf16)b.w;
  *(uint4*)(v + i * 8) = o.u;
}

// ---------------------------------------------------------------- shared MMA step
__device__ __forceinline__ void mma_step_rowrow(const __bf16* Abuf, const __bf16* Bbuf,
                                                int wm, int wn, int hf, int l16,
                                                f32x8 acc[2][4]) {
  FragCvt afr[2], bfr[4];
#pragma unroll
  for (int mt = 0; mt < 2; ++mt) {
    const __bf16* rp = Abuf + (wm * 32 + mt * 16 + l16) * A_ST;
    afr[mt].u[0] = *(const uint4*)(rp + hf * 8);        // K = 8h + 0..7
    afr[mt].u[1] = *(const uint4*)(rp + 16 + hf * 8);   // K = 16 + 8h + 0..7
  }
#pragma unroll
  for (int nt = 0; nt < 4; ++nt) {
    const __bf16* rp = Bbuf + (wn * 64 + nt * 16 + l16) * A_ST;
    bfr[nt].u[0] = *(const uint4*)(rp + hf * 8);        // K = 8h + 0..7
    bfr[nt].u[1] = *(const uint4*)(rp + 16 + hf * 8);   // K = 16 + 8h + 0..7
  }
#pragma unroll
  for (int mt = 0; mt < 2; ++mt)
#pragma unroll
    for (int nt = 0; nt < 4; ++nt)
      acc[mt][nt] = __builtin_amdgcn_wmma_f32_16x16x32_bf16(
          false, afr[mt].v, false, bfr[nt].v, (short)0, acc[mt][nt], false, false);
}

// ---------------------------------------------------------------- energy = v * v^T
__global__ __launch_bounds__(256)
void cam_gemm_energy(const __bf16* __restrict__ v, float* __restrict__ energy) {
  __shared__ __bf16 As[2][TILE * A_ST];
  __shared__ __bf16 Bs[2][TILE * A_ST];

  const int b  = blockIdx.z;
  const int m0 = blockIdx.x * TILE;
  const int n0 = blockIdx.y * TILE;
  const __bf16* vb = v + (size_t)b * CCH * NSP;

  const int tid  = threadIdx.x;
  const int lane = tid & 31;
  const int wave = tid >> 5;
  const int hf   = lane >> 4;
  const int l16  = lane & 15;
  const int wm   = wave >> 1;   // 0..3 : 32-row band
  const int wn   = wave & 1;    // 0..1 : 64-col band

  f32x8 acc[2][4];
#pragma unroll
  for (int mt = 0; mt < 2; ++mt)
#pragma unroll
    for (int nt = 0; nt < 4; ++nt) acc[mt][nt] = {};

#if USE_TDM
  const unsigned ldsA[2] = { (unsigned)(size_t)(void*)&As[0][0], (unsigned)(size_t)(void*)&As[1][0] };
  const unsigned ldsB[2] = { (unsigned)(size_t)(void*)&Bs[0][0], (unsigned)(size_t)(void*)&Bs[1][0] };
  if (wave == 0) {
    tdm_load_tile_2d(vb + (size_t)m0 * NSP, ldsA[0], KSTEP, TILE, NSP);
    tdm_load_tile_2d(vb + (size_t)n0 * NSP, ldsB[0], KSTEP, TILE, NSP);
    __builtin_amdgcn_s_wait_tensorcnt(0);
  }
  __syncthreads();
  int buf = 0;
  for (int kb = 0; kb < NSP; kb += KSTEP) {
    if (wave == 0 && (kb + KSTEP) < NSP) {      // prefetch next tile into other buffer
      tdm_load_tile_2d(vb + (size_t)m0 * NSP + (kb + KSTEP), ldsA[buf ^ 1], KSTEP, TILE, NSP);
      tdm_load_tile_2d(vb + (size_t)n0 * NSP + (kb + KSTEP), ldsB[buf ^ 1], KSTEP, TILE, NSP);
    }
    mma_step_rowrow(&As[buf][0], &Bs[buf][0], wm, wn, hf, l16, acc);
    if (wave == 0) __builtin_amdgcn_s_wait_tensorcnt(0);
    __syncthreads();
    buf ^= 1;
  }
#else
  const int sr  = tid >> 2;
  const int seg = tid & 3;
  for (int kb = 0; kb < NSP; kb += KSTEP) {
    uint4 a0 = *(const uint4*)(vb + (size_t)(m0 + sr)      * NSP + kb + seg * 8);
    uint4 a1 = *(const uint4*)(vb + (size_t)(m0 + sr + 64) * NSP + kb + seg * 8);
    uint4 b0 = *(const uint4*)(vb + (size_t)(n0 + sr)      * NSP + kb + seg * 8);
    uint4 b1 = *(const uint4*)(vb + (size_t)(n0 + sr + 64) * NSP + kb + seg * 8);
    *(uint4*)(&As[0][0] + (sr)      * A_ST + seg * 8) = a0;
    *(uint4*)(&As[0][0] + (sr + 64) * A_ST + seg * 8) = a1;
    *(uint4*)(&Bs[0][0] + (sr)      * A_ST + seg * 8) = b0;
    *(uint4*)(&Bs[0][0] + (sr + 64) * A_ST + seg * 8) = b1;
    __syncthreads();
    mma_step_rowrow(&As[0][0], &Bs[0][0], wm, wn, hf, l16, acc);
    __syncthreads();
  }
#endif

  float* eb = energy + (size_t)b * CCH * CCH;
#pragma unroll
  for (int mt = 0; mt < 2; ++mt)
#pragma unroll
    for (int nt = 0; nt < 4; ++nt) {
      const int col = n0 + wn * 64 + nt * 16 + l16;
#pragma unroll
      for (int r = 0; r < 8; ++r) {
        const int row = m0 + wm * 32 + mt * 16 + hf * 8 + r;
        eb[(size_t)row * CCH + col] = acc[mt][nt][r];
      }
    }
}

// ---------------------------------------------------------------- inverted softmax
// softmax(max(e) - e) == exp(min(e) - e) / sum(exp(min(e) - e))
__global__ __launch_bounds__(256)
void cam_softmax(const float* __restrict__ energy, __bf16* __restrict__ att) {
  __shared__ float red[256];
  const int row = blockIdx.x;
  const int t   = threadIdx.x;
  const float* e = energy + (size_t)row * CCH;
  float a0 = e[t];
  float a1 = e[t + 256];
  red[t] = fminf(a0, a1);
  __syncthreads();
#pragma unroll
  for (int s = 128; s > 0; s >>= 1) {
    if (t < s) red[t] = fminf(red[t], red[t + s]);
    __syncthreads();
  }
  const float mn = red[0];
  __syncthreads();
  float s0 = __expf(mn - a0);
  float s1 = __expf(mn - a1);
  red[t] = s0 + s1;
  __syncthreads();
#pragma unroll
  for (int s = 128; s > 0; s >>= 1) {
    if (t < s) red[t] += red[t + s];
    __syncthreads();
  }
  const float inv = 1.0f / red[0];
  __bf16* ar = att + (size_t)row * CCH;
  ar[t]       = (__bf16)(s0 * inv);
  ar[t + 256] = (__bf16)(s1 * inv);
}

// ---------------------------------------------------------------- out = gamma*(att*v) + x
__global__ __launch_bounds__(256)
void cam_gemm_out(const __bf16* __restrict__ att, const __bf16* __restrict__ v,
                  const float* __restrict__ x, const float* __restrict__ gamma,
                  float* __restrict__ out) {
  __shared__ __bf16        As[TILE * A_ST];
  __shared__ unsigned int  Bt[TILE * BT_ST];   // v tile transposed: [n][k-pair dwords]

  const int b  = blockIdx.z;
  const int m0 = blockIdx.x * TILE;   // channel rows
  const int n0 = blockIdx.y * TILE;   // spatial cols
  const __bf16* ab = att + (size_t)b * CCH * CCH;
  const __bf16* vb = v   + (size_t)b * CCH * NSP;

  const int tid  = threadIdx.x;
  const int lane = tid & 31;
  const int wave = tid >> 5;
  const int hf   = lane >> 4;
  const int l16  = lane & 15;
  const int wm   = wave >> 1;
  const int wn   = wave & 1;

  f32x8 acc[2][4];
#pragma unroll
  for (int mt = 0; mt < 2; ++mt)
#pragma unroll
    for (int nt = 0; nt < 4; ++nt) acc[mt][nt] = {};

  const int p = tid & 15;   // B staging: k-pair index (k = 2p, 2p+1)
  const int q = tid >> 4;   // B staging: 8-wide n segment
#if USE_TDM
  const unsigned ldsA = (unsigned)(size_t)(void*)&As[0];
#else
  const int sr  = tid >> 2;
  const int seg = tid & 3;
#endif

  for (int kb = 0; kb < CCH; kb += KSTEP) {
#if USE_TDM
    if (wave == 0)   // A tile (att rows m0..m0+127, k kb..kb+31) via Tensor Data Mover
      tdm_load_tile_2d(ab + (size_t)m0 * CCH + kb, ldsA, KSTEP, TILE, CCH);
#else
    uint4 a0 = *(const uint4*)(ab + (size_t)(m0 + sr)      * CCH + kb + seg * 8);
    uint4 a1 = *(const uint4*)(ab + (size_t)(m0 + sr + 64) * CCH + kb + seg * 8);
    *(uint4*)(As + (sr)      * A_ST + seg * 8) = a0;
    *(uint4*)(As + (sr + 64) * A_ST + seg * 8) = a1;
#endif
    // B tile: transpose v[kb..kb+32)[n0..n0+128) into n-major, k-pairs packed per dword
    u16x8 ev = *(const u16x8*)(vb + (size_t)(kb + 2 * p)     * NSP + n0 + q * 8);
    u16x8 ov = *(const u16x8*)(vb + (size_t)(kb + 2 * p + 1) * NSP + n0 + q * 8);
#pragma unroll
    for (int j = 0; j < 8; ++j)
      Bt[(q * 8 + j) * BT_ST + p] = (unsigned)ev[j] | ((unsigned)ov[j] << 16);
#if USE_TDM
    if (wave == 0) __builtin_amdgcn_s_wait_tensorcnt(0);
#endif
    __syncthreads();

    FragCvt afr[2], bfr[4];
#pragma unroll
    for (int mt = 0; mt < 2; ++mt) {
      const __bf16* rp = As + (wm * 32 + mt * 16 + l16) * A_ST;
      afr[mt].u[0] = *(const uint4*)(rp + hf * 8);
      afr[mt].u[1] = *(const uint4*)(rp + 16 + hf * 8);
    }
#pragma unroll
    for (int nt = 0; nt < 4; ++nt) {
      const unsigned int* rp = Bt + (wn * 64 + nt * 16 + l16) * BT_ST + hf * 4;
      bfr[nt].u[0] = *(const uint4*)(rp);        // K = 8h + 0..7
      bfr[nt].u[1] = *(const uint4*)(rp + 8);    // K = 16 + 8h + 0..7
    }
#pragma unroll
    for (int mt = 0; mt < 2; ++mt)
#pragma unroll
      for (int nt = 0; nt < 4; ++nt)
        acc[mt][nt] = __builtin_amdgcn_wmma_f32_16x16x32_bf16(
            false, afr[mt].v, false, bfr[nt].v, (short)0, acc[mt][nt], false, false);
    __syncthreads();
  }

  const float g = gamma[0];
  const float* xb = x   + (size_t)b * CCH * NSP;
  float*       ob = out + (size_t)b * CCH * NSP;
#pragma unroll
  for (int mt = 0; mt < 2; ++mt)
#pragma unroll
    for (int nt = 0; nt < 4; ++nt) {
      const int col = n0 + wn * 64 + nt * 16 + l16;
#pragma unroll
      for (int r = 0; r < 8; ++r) {
        const int row = m0 + wm * 32 + mt * 16 + hf * 8 + r;
        const size_t idx = (size_t)row * NSP + col;
        ob[idx] = g * acc[mt][nt][r] + xb[idx];
      }
    }
}

// ---------------------------------------------------------------- launch
extern "C" void kernel_launch(void* const* d_in, const int* in_sizes, int n_in,
                              void* d_out, int out_size, void* d_ws, size_t ws_size,
                              hipStream_t stream) {
  (void)in_sizes; (void)n_in; (void)out_size; (void)ws_size;
  const float* x     = (const float*)d_in[0];
  const float* gamma = (const float*)d_in[1];
  float*       out   = (float*)d_out;

  const size_t VBF_BYTES = (size_t)BATCH * CCH * NSP * 2;  // 64 MiB bf16 copy of x
  const size_t EN_BYTES  = (size_t)BATCH * CCH * CCH * 4;  // 16 MiB energy fp32
  __bf16* v_bf   = (__bf16*)d_ws;
  float*  energy = (float*)((char*)d_ws + VBF_BYTES);
  __bf16* att    = (__bf16*)((char*)d_ws + VBF_BYTES + EN_BYTES);

  const int cvt_blocks = (BATCH * CCH * NSP / 8) / 256;    // 16384
  cam_convert<<<cvt_blocks, 256, 0, stream>>>(x, v_bf);
  cam_gemm_energy<<<dim3(CCH / TILE, CCH / TILE, BATCH), 256, 0, stream>>>(v_bf, energy);
  cam_softmax<<<BATCH * CCH, 256, 0, stream>>>(energy, att);
  cam_gemm_out<<<dim3(CCH / TILE, NSP / TILE, BATCH), 256, 0, stream>>>(att, v_bf, x, gamma, out);
}
